// ISI_loss_function_52793738003265
// MI455X (gfx1250) — matrix operand order repaired and checked by reading, compiled, not run on hardware
//
#include <hip/hip_runtime.h>
#include <hip/hip_bf16.h>

typedef __attribute__((ext_vector_type(16))) _Float16 v16h;
typedef __attribute__((ext_vector_type(8)))  _Float16 v8h;
typedef __attribute__((ext_vector_type(8)))  float    v8f;

#define B_SZ   1024
#define NP     2048
#define MID    1024
#define MU     8
#define TT     128      // T
#define K2     256      // 2T
#define NFFT   8548     // NP + PAD
#define NBW    4989
#define BIN0   12
#define NTILES 312      // ceil(4989/16)

// ws layout: floats [0..259) = accumulators; byte 2048+: f16 copy of dnn_out
#define WS_BW     0
#define WS_SYM    1
#define WS_POW    2
#define WS_PD     3
#define WS_TOTAL  (3 + K2)
#define WS_F16_OFF 2048
#define WS_NEED   ((size_t)WS_F16_OFF + (size_t)B_SZ * NP * 2)

__global__ void k_init(float* __restrict__ ws) {
    int i = blockIdx.x * blockDim.x + threadIdx.x;
    if (i < WS_TOTAL) ws[i] = 0.0f;
}

// ---------------- dnn_out f32 -> f16 pack (once; reused by all N-tiles) -----
__global__ __launch_bounds__(256) void k_pack(const float* __restrict__ dnn,
                                              _Float16* __restrict__ out) {
    long i = (long)(blockIdx.x * 256 + threadIdx.x) * 8;
    float4 a = *(const float4*)(dnn + i);
    float4 b = *(const float4*)(dnn + i + 4);
    v8h h;
    h[0] = (_Float16)a.x; h[1] = (_Float16)a.y;
    h[2] = (_Float16)a.z; h[3] = (_Float16)a.w;
    h[4] = (_Float16)b.x; h[5] = (_Float16)b.y;
    h[6] = (_Float16)b.z; h[7] = (_Float16)b.w;
    *(v8h*)(out + i) = h;
}

// ---------------- Kernel A: ISI main-loss term (per-batch gathered matvec) --
__global__ __launch_bounds__(256) void k_isi(const float* __restrict__ dnn,
                                             const float* __restrict__ batch,
                                             const float* __restrict__ isr,
                                             const float* __restrict__ isim,
                                             const float* __restrict__ ich,
                                             const float* __restrict__ noise,
                                             float* __restrict__ ws) {
    __shared__ float sp[NP];
    __shared__ float swr[K2];
    __shared__ float swi[K2];
    const int b = blockIdx.x;
    const int t = threadIdx.x;
    const int bi = (int)batch[b * 4 + 0];
    const float h  = batch[b * 4 + 1];
    const float xr = batch[b * 4 + 2];
    const float xi = batch[b * 4 + 3];
    const float* row = dnn + (long)bi * NP;
    for (int j = t; j < NP; j += 256) sp[j] = row[j];
    {
        float c = ich[b * K2 + t];
        swr[t] = c * isr[b * K2 + t];
        swi[t] = c * isim[b * K2 + t];
    }
    __syncthreads();
    const int d  = t;
    const int st = MID - TT + d;               // 896 + d
    const float pm = sp[st];
    float yr = h * xr * pm;
    float yi = h * xi * pm;
#pragma unroll 8
    for (int k = 0; k < K2; k++) {
        int kk  = (k < TT) ? (k - TT) : (k - TT + 1);   // -128..-1, 1..128
        int idx = (st + kk * MU + NP) & (NP - 1);
        float pv = sp[idx];
        yr += swr[k] * pv;
        yi += swi[k] * pv;
    }
    const float nz = noise[b];
    const float er = yr + nz - xr;
    const float ei = yi + nz - xi;
    atomicAdd(&ws[WS_PD + d], er * er + ei * ei);
}

// ---------------- Kernel B: symmetry + power losses -------------------------
__global__ __launch_bounds__(256) void k_sym_pow(const float* __restrict__ dnn,
                                                 const float* __restrict__ pul_power,
                                                 float* __restrict__ ws) {
    __shared__ float r1[256];
    __shared__ float r2[256];
    const int b = blockIdx.x, t = threadIdx.x;
    const float* row = dnn + (long)b * NP;
    float s_sym = 0.0f, s_pow = 0.0f;
    for (int j = t; j < MID; j += 256) {
        float dsym = row[j] - row[NP - 1 - j];
        s_sym += dsym * dsym;
    }
    for (int j = t; j < NP; j += 256) {
        float v = row[j];
        s_pow += v * v;
    }
    r1[t] = s_sym; r2[t] = s_pow;
    __syncthreads();
    for (int s = 128; s > 0; s >>= 1) {
        if (t < s) { r1[t] += r1[t + s]; r2[t] += r2[t + s]; }
        __syncthreads();
    }
    if (t == 0) {
        atomicAdd(&ws[WS_SYM], r1[0]);
        float pd = r2[0] / (float)NP - pul_power[0];
        atomicAdd(&ws[WS_POW], pd * pd);
    }
}

// ---------------- Kernel C: DFT-as-GEMM bandwidth loss (WMMA f16) -----------
// grid = (NTILES, 2), block = 256 (8 waves). Block y covers 512 rows; each
// wave handles 4 M-tiles of 16 rows. Basis tile stored n-major/k-contiguous
// so each lane fetches its B fragment with 2x ds_load_b128 per matrix.
template <bool PRE>
__global__ __launch_bounds__(256) void k_dft(const float* __restrict__ dnn,
                                             const _Float16* __restrict__ aF16,
                                             const float* __restrict__ freq_resp,
                                             float* __restrict__ ws) {
    // packed half2 per entry: bcp[n*16 + k2] holds (cos(k=2k2), cos(k=2k2+1))
    __shared__ unsigned int bcp[16 * 16];
    __shared__ unsigned int bsp[16 * 16];
    __shared__ float red[256];

    const int t    = threadIdx.x;
    const int lane = t & 31;
    const int wv   = t >> 5;
    const int n0   = blockIdx.x * 16;
    const int half = lane >> 4;
    const int nn   = lane & 15;
    const int row_base = blockIdx.y * 512 + wv * 64;

    v8f cre[4] = {};
    v8f cim[4] = {};

    // basis writer mapping: consecutive threads -> consecutive LDS dwords
    const int wk2 = t & 15;        // k-pair index: k = 2*wk2, 2*wk2+1
    const int wn  = t >> 4;        // column 0..15
    const int wf  = BIN0 + n0 + wn;
    const float twofac = -6.28318530717958647692f / (float)NFFT;

    // per-lane B fragment source: halves [nn*32 + 8*half .. +7] and [nn*32 + 16 + 8*half .. +7]
    const _Float16* bcl = (const _Float16*)bcp + nn * 32 + 8 * half;
    const _Float16* bsl = (const _Float16*)bsp + nn * 32 + 8 * half;

    // per-tile A row pointers (lane holds row M = nn of each tile)
    const float*    apf[4];
    const _Float16* aph[4];
#pragma unroll
    for (int i = 0; i < 4; i++) {
        long m = row_base + i * 16 + nn;
        apf[i] = dnn  + m * NP;
        aph[i] = aF16 + m * NP;
    }

    for (int k0 = 0; k0 < NP; k0 += 32) {
        __syncthreads();
        {   // fill basis tile: each thread does one k-pair at one column
            int p0 = (wf * (k0 + 2 * wk2)) % NFFT;
            int p1 = p0 + wf; if (p1 >= NFFT) p1 -= NFFT;
            float s0, c0, s1, c1;
            __sincosf(twofac * (float)p0, &s0, &c0);
            __sincosf(twofac * (float)p1, &s1, &c1);
            union { unsigned int u; _Float16 h[2]; } pc, ps;
            pc.h[0] = (_Float16)c0; pc.h[1] = (_Float16)c1;
            ps.h[0] = (_Float16)s0; ps.h[1] = (_Float16)s1;
            bcp[wn * 16 + wk2] = pc.u;
            bsp[wn * 16 + wk2] = ps.u;
        }
        __syncthreads();

        v8h bc0 = *(const v8h*)(bcl);
        v8h bc1 = *(const v8h*)(bcl + 16);
        v8h bs0 = *(const v8h*)(bsl);
        v8h bs1 = *(const v8h*)(bsl + 16);
        v16h bcv = __builtin_shufflevector(bc0, bc1, 0, 1, 2, 3, 4, 5, 6, 7,
                                           8, 9, 10, 11, 12, 13, 14, 15);
        v16h bsv = __builtin_shufflevector(bs0, bs1, 0, 1, 2, 3, 4, 5, 6, 7,
                                           8, 9, 10, 11, 12, 13, 14, 15);

#pragma unroll
        for (int i = 0; i < 4; i++) {
            v16h av;
            if constexpr (PRE) {
                v8h lo = *(const v8h*)(aph[i] + k0 + 8 * half);
                v8h hi = *(const v8h*)(aph[i] + k0 + 16 + 8 * half);
                av = __builtin_shufflevector(lo, hi, 0, 1, 2, 3, 4, 5, 6, 7,
                                             8, 9, 10, 11, 12, 13, 14, 15);
            } else {
#pragma unroll
                for (int j = 0; j < 8; j++) {
                    int kA = (j < 4) ? (8 * half + 2 * j)
                                     : (16 + 8 * half + 2 * (j - 4));
                    float2 p2 = *(const float2*)(apf[i] + k0 + kA);
                    av[2 * j + 0] = (_Float16)p2.x;
                    av[2 * j + 1] = (_Float16)p2.y;
                }
            }
            cre[i] = __builtin_amdgcn_wmma_f32_16x16x32_f16(
                false, av, false, bcv, (short)0, cre[i], false, false);
            cim[i] = __builtin_amdgcn_wmma_f32_16x16x32_f16(
                false, av, false, bsv, (short)0, cim[i], false, false);
        }
    }

    // bandwidth-loss partial: mag = sqrt(re^2+im^2); sum (mag - freq_resp)^2
    float acc = 0.0f;
    const int bin = n0 + nn;
    if (bin < NBW) {
        const float fr = freq_resp[bin];
#pragma unroll
        for (int i = 0; i < 4; i++) {
#pragma unroll
            for (int v = 0; v < 8; v++) {
                float re  = cre[i][v];
                float im  = cim[i][v];
                float mag = sqrtf(re * re + im * im);
                float df  = mag - fr;
                acc += df * df;
            }
        }
    }
    red[t] = acc;
    __syncthreads();
    for (int s = 128; s > 0; s >>= 1) {
        if (t < s) red[t] += red[t + s];
        __syncthreads();
    }
    if (t == 0) atomicAdd(&ws[WS_BW], red[0]);
}

// ---------------- Final combine ---------------------------------------------
__global__ void k_final(const float* __restrict__ prob,
                        const float* __restrict__ ws,
                        const float* __restrict__ M_loss,
                        const float* __restrict__ M_sym,
                        const float* __restrict__ M_power,
                        const float* __restrict__ M_bandwidth,
                        float* __restrict__ out) {
    if (threadIdx.x == 0 && blockIdx.x == 0) {
        float lm = 0.0f;
        for (int d = 0; d < K2; d++)
            lm += prob[MID - TT + d] * (ws[WS_PD + d] / (float)B_SZ);
        float loss = lm * M_loss[0]
                   + M_sym[0]       * ws[WS_SYM] / ((float)B_SZ * (float)MID)
                   + M_power[0]     * ws[WS_POW] / (float)B_SZ
                   + M_bandwidth[0] * ws[WS_BW]  / ((float)B_SZ * (float)NBW);
        out[0] = loss;
    }
}

extern "C" void kernel_launch(void* const* d_in, const int* in_sizes, int n_in,
                              void* d_out, int out_size, void* d_ws, size_t ws_size,
                              hipStream_t stream) {
    const float* dnn       = (const float*)d_in[1];
    const float* batch     = (const float*)d_in[3];
    const float* isr       = (const float*)d_in[4];
    const float* isim      = (const float*)d_in[5];
    const float* ich       = (const float*)d_in[6];
    const float* noise     = (const float*)d_in[8];
    const float* prob      = (const float*)d_in[9];
    const float* M_loss    = (const float*)d_in[10];
    const float* M_sym     = (const float*)d_in[11];
    const float* M_power   = (const float*)d_in[12];
    const float* M_bw      = (const float*)d_in[13];
    const float* pul_power = (const float*)d_in[14];
    const float* freq_resp = (const float*)d_in[15];
    float*     ws   = (float*)d_ws;
    float*     out  = (float*)d_out;
    _Float16*  aF16 = (_Float16*)((char*)d_ws + WS_F16_OFF);
    const bool pre  = (ws_size >= WS_NEED);

    k_init<<<1, 512, 0, stream>>>(ws);
    if (pre) k_pack<<<(B_SZ * NP) / (256 * 8), 256, 0, stream>>>(dnn, aF16);
    k_isi<<<B_SZ, 256, 0, stream>>>(dnn, batch, isr, isim, ich, noise, ws);
    k_sym_pow<<<B_SZ, 256, 0, stream>>>(dnn, pul_power, ws);
    if (pre)
        k_dft<true><<<dim3(NTILES, 2), 256, 0, stream>>>(dnn, aF16, freq_resp, ws);
    else
        k_dft<false><<<dim3(NTILES, 2), 256, 0, stream>>>(dnn, aF16, freq_resp, ws);
    k_final<<<1, 1, 0, stream>>>(prob, ws, M_loss, M_sym, M_power, M_bw, out);
}
